// UniShapeModel_72730976190820
// MI455X (gfx1250) — matrix operand order repaired and checked by reading, not compiled, not run on hardware
//
#include <hip/hip_runtime.h>

#define B_SZ   1024
#define N_TOK  512
#define D_DIM  128
#define H_DIM  256
#define C_CLS  50
#define K_TOP  128
#define CP     64           // classes padded to 4 WMMA n-tiles
#define TAU_INV 2.0f
#define ALPHA   0.01f
#define MOM     0.9f
#define BN_INV  0.9999950000374997f   // 1/sqrt(1+1e-5)
#define WAVES   4

typedef __attribute__((ext_vector_type(16))) __bf16 v16bf;
typedef __attribute__((ext_vector_type(8)))  float  v8f;

union BFrag { v16bf v; unsigned int u[8]; };

#if defined(__has_builtin)
#if __has_builtin(__builtin_amdgcn_cvt_pk_bf16_f32)
#define HAVE_CVT_PK_BF16 1
#endif
#if __has_builtin(__builtin_amdgcn_perm)
#define HAVE_PERM 1
#endif
#endif

// pack two f32 -> packed bf16 pair (lo in [15:0], hi in [31:16])
__device__ __forceinline__ unsigned int pack2bf(float lo, float hi) {
#if defined(HAVE_CVT_PK_BF16)
  typedef __attribute__((ext_vector_type(2))) __bf16 v2bf;
  v2bf r = __builtin_amdgcn_cvt_pk_bf16_f32(lo, hi);
  return __builtin_bit_cast(unsigned int, r);
#elif defined(HAVE_PERM)
  // single v_perm_b32: take high 16 bits of each f32 (truncating bf16)
  return __builtin_amdgcn_perm(__float_as_uint(hi), __float_as_uint(lo),
                               0x07060302u);
#else
  unsigned int a = __float_as_uint(lo), b = __float_as_uint(hi);
  a = (a + 0x7FFFu + ((a >> 16) & 1u)) >> 16;
  b = (b + 0x7FFFu + ((b >> 16) & 1u)) >> 16;
  return (a & 0xFFFFu) | (b << 16);
#endif
}
__device__ __forceinline__ unsigned short f32bf(float f) {
#if defined(HAVE_CVT_PK_BF16)
  return (unsigned short)(pack2bf(f, 0.f) & 0xFFFFu);
#else
  return (unsigned short)(__float_as_uint(f) >> 16);  // truncating bf16
#endif
}
__device__ __forceinline__ v8f wmma_bf16(const BFrag& A, const BFrag& Bm, v8f C) {
  return __builtin_amdgcn_wmma_f32_16x16x32_bf16(false, A.v, false, Bm.v,
                                                 (short)0, C, false, false);
}
// A-fragment (16x32 bf16) from a global f32 row pointer, K-chunk c.
__device__ __forceinline__ BFrag afrag_g(const float* xr, int c, int hf) {
  BFrag f;
#pragma unroll
  for (int v = 0; v < 8; ++v) {
    int k = c * 32 + hf * 16 + 2 * v;
    float2 p = *reinterpret_cast<const float2*>(xr + k);
    f.u[v] = pack2bf(p.x, p.y);
  }
  return f;
}
// B-fragment (32x16 bf16) from global f32 weight W[K][ld], column n (guarded)
__device__ __forceinline__ BFrag bfrag_g(const float* W, int ld, int n, int ncols,
                                         int c, int hf) {
  BFrag f;
  int kb = c * 32 + hf * 16;
#pragma unroll
  for (int v = 0; v < 8; ++v) {
    int k = kb + 2 * v;
    float lo = (n < ncols) ? W[(size_t)k * ld + n] : 0.f;
    float hi = (n < ncols) ? W[(size_t)(k + 1) * ld + n] : 0.f;
    f.u[v] = pack2bf(lo, hi);
  }
  return f;
}
// B-fragment from LDS in fragment order: two b128 loads, contiguous VGPRs
__device__ __forceinline__ BFrag bfrag_lds(const unsigned int* base) {
  const uint4* p = reinterpret_cast<const uint4*>(base);
  uint4 q0 = p[0], q1 = p[1];
  BFrag f;
  f.u[0] = q0.x; f.u[1] = q0.y; f.u[2] = q0.z; f.u[3] = q0.w;
  f.u[4] = q1.x; f.u[5] = q1.y; f.u[6] = q1.z; f.u[7] = q1.w;
  return f;
}

// ---------------------------------------------------------------------------
// Kernel 1: token MLP (Linear->BN->ReLU->Linear) + attention score, fused.
// One wave = 16-token tile. Weights staged in LDS in WMMA B-fragment order
// (8 packed bf16-pair dwords per (chunk, half, col) contiguous -> b128 loads).
// ---------------------------------------------------------------------------
__global__ __launch_bounds__(128) void token_mlp_kernel(
    const float* __restrict__ x, const float* __restrict__ w1,
    const float* __restrict__ b1, const float* __restrict__ g1,
    const float* __restrict__ be1, const float* __restrict__ w2,
    const float* __restrict__ b2, const float* __restrict__ wa1,
    const float* __restrict__ ba1, const float* __restrict__ wa2,
    const float* __restrict__ ba2, float* __restrict__ st,
    float* __restrict__ scores) {
  extern __shared__ unsigned char smem_raw[];
  unsigned int*   w1p = (unsigned int*)smem_raw;            // frag-order (64KB)
  unsigned int*   w2p = w1p + 64 * 256;                     // frag-order (64KB)
  float*          wa1s = (float*)(w2p + 128 * 128);         // [128*8] f32 (4KB)
  unsigned short* hball = (unsigned short*)(wa1s + 1024);   // WAVES*[16][256] (32KB)

  int tid = threadIdx.x;
  // w1 fragment-order: idx = c*4096 + n*16 + hf*8 + v ; k = c*32 + hf*16 + 2v
  for (int i = tid; i < 4 * 4096; i += 128) {
    int v = i & 7, hfb = (i >> 3) & 1, n = (i >> 4) & 255, c = i >> 12;
    int k = c * 32 + hfb * 16 + 2 * v;
    w1p[i] = pack2bf(w1[k * H_DIM + n], w1[(k + 1) * H_DIM + n]);
  }
  // w2 fragment-order: idx = c*2048 + n*16 + hf*8 + v
  for (int i = tid; i < 8 * 2048; i += 128) {
    int v = i & 7, hfb = (i >> 3) & 1, n = (i >> 4) & 127, c = i >> 11;
    int k = c * 32 + hfb * 16 + 2 * v;
    w2p[i] = pack2bf(w2[k * D_DIM + n], w2[(k + 1) * D_DIM + n]);
  }
  for (int i = tid; i < 1024; i += 128) wa1s[i] = wa1[i];
  __syncthreads();

  int wv = tid >> 5, lane = tid & 31, lm = lane & 15, hf = lane >> 4;
  size_t row0 = ((size_t)blockIdx.x * WAVES + wv) * 16;
  const float* xr = x + (row0 + lm) * D_DIM;
  unsigned short* hb = hball + wv * (16 * 256);

  BFrag A[4];
#pragma unroll
  for (int c = 0; c < 4; ++c) A[c] = afrag_g(xr, c, hf);

  // GEMM1: h = relu(bn(x@w1 + b1))  -> LDS as bf16 [16][256]
#pragma unroll 4
  for (int t = 0; t < 16; ++t) {
    int n = t * 16 + lm;
    v8f acc = {0.f, 0.f, 0.f, 0.f, 0.f, 0.f, 0.f, 0.f};
#pragma unroll
    for (int c = 0; c < 4; ++c)
      acc = wmma_bf16(A[c], bfrag_lds(w1p + c * 4096 + n * 16 + hf * 8), acc);
    float bb = b1[n], gg = g1[n], ee = be1[n];
#pragma unroll
    for (int v = 0; v < 8; ++v) {
      float h = fmaxf(gg * ((acc[v] + bb) * BN_INV) + ee, 0.f);
      hb[(hf * 8 + v) * H_DIM + n] = f32bf(h);
    }
  }
  __syncthreads();

  // GEMM2: st = h@w2 + b2. A-frags of h straight from LDS (pairs packed).
  BFrag A2[8];
#pragma unroll
  for (int c = 0; c < 8; ++c)
#pragma unroll
    for (int v = 0; v < 8; ++v) {
      int k = c * 32 + hf * 16 + 2 * v;
      A2[c].u[v] = *reinterpret_cast<const unsigned int*>(hb + lm * H_DIM + k);
    }
  float* stb = reinterpret_cast<float*>(hb);  // reuse as [16][128] f32
#pragma unroll 2
  for (int t = 0; t < 8; ++t) {
    int n = t * 16 + lm;
    v8f acc = {0.f, 0.f, 0.f, 0.f, 0.f, 0.f, 0.f, 0.f};
#pragma unroll
    for (int c = 0; c < 8; ++c)
      acc = wmma_bf16(A2[c], bfrag_lds(w2p + c * 2048 + n * 16 + hf * 8), acc);
    float bb = b2[n];
#pragma unroll
    for (int v = 0; v < 8; ++v)
      stb[(hf * 8 + v) * D_DIM + n] = acc[v] + bb;
  }
  // coalesced b128 store of the st tile (16 rows x 128 f32)
  for (int i = lane; i < 16 * 32; i += 32) {
    int m = i >> 5, q = i & 31;
    float4 val = reinterpret_cast<const float4*>(stb + m * D_DIM)[q];
    reinterpret_cast<float4*>(st + (row0 + m) * D_DIM)[q] = val;
  }
  __syncthreads();

  // attention head: sigmoid(tanh(st@wa1+ba1)@wa2+ba2), one row per lane 0-15
  if (lane < 16) {
    const float* srow = stb + lm * D_DIM;
    float aj[8];
#pragma unroll
    for (int j = 0; j < 8; ++j) aj[j] = ba1[j];
    for (int kk = 0; kk < D_DIM; ++kk) {
      float xv = srow[kk];
#pragma unroll
      for (int j = 0; j < 8; ++j) aj[j] += xv * wa1s[kk * 8 + j];
    }
    float z = ba2[0];
#pragma unroll
    for (int j = 0; j < 8; ++j) z += tanhf(aj[j]) * wa2[j];
    scores[row0 + lm] = 1.f / (1.f + expf(-z));
  }
}

// ---------------------------------------------------------------------------
// Kernel 2: class-token fc2 head (used for both features and cls_results)
// ---------------------------------------------------------------------------
__global__ __launch_bounds__(128) void head_kernel(
    const float* __restrict__ xin, const float* __restrict__ wA,
    const float* __restrict__ bA, const float* __restrict__ gA,
    const float* __restrict__ eA, const float* __restrict__ wB,
    const float* __restrict__ bB, float* __restrict__ out, int out_ld, int ncols,
    int ntiles2) {
  __shared__ unsigned short hbuf[WAVES][16 * H_DIM];
  int tid = threadIdx.x, wv = tid >> 5, lane = tid & 31, lm = lane & 15,
      hf = lane >> 4;
  size_t row0 = ((size_t)blockIdx.x * WAVES + wv) * 16;
  const float* xr = xin + (row0 + lm) * D_DIM;
  unsigned short* hb = hbuf[wv];

  BFrag A[4];
#pragma unroll
  for (int c = 0; c < 4; ++c) A[c] = afrag_g(xr, c, hf);

  for (int t = 0; t < 16; ++t) {
    int n = t * 16 + lm;
    v8f acc = {0.f, 0.f, 0.f, 0.f, 0.f, 0.f, 0.f, 0.f};
#pragma unroll
    for (int c = 0; c < 4; ++c)
      acc = wmma_bf16(A[c], bfrag_g(wA, H_DIM, n, H_DIM, c, hf), acc);
    float bb = bA[n], gg = gA[n], ee = eA[n];
#pragma unroll
    for (int v = 0; v < 8; ++v) {
      float h = fmaxf(gg * ((acc[v] + bb) * BN_INV) + ee, 0.f);
      hb[(hf * 8 + v) * H_DIM + n] = f32bf(h);
    }
  }
  __syncthreads();

  BFrag A2[8];
#pragma unroll
  for (int c = 0; c < 8; ++c)
#pragma unroll
    for (int v = 0; v < 8; ++v) {
      int k = c * 32 + hf * 16 + 2 * v;
      A2[c].u[v] = *reinterpret_cast<const unsigned int*>(hb + lm * H_DIM + k);
    }
  for (int t = 0; t < ntiles2; ++t) {
    int n = t * 16 + lm;
    v8f acc = {0.f, 0.f, 0.f, 0.f, 0.f, 0.f, 0.f, 0.f};
#pragma unroll
    for (int c = 0; c < 8; ++c)
      acc = wmma_bf16(A2[c], bfrag_g(wB, ncols, n, ncols, c, hf), acc);
    if (n < ncols) {
      float bb = bB[n];
#pragma unroll
      for (int v = 0; v < 8; ++v)
        out[(row0 + hf * 8 + v) * out_ld + n] = acc[v] + bb;
    }
  }
}

// ---------------------------------------------------------------------------
// Small kernels
// ---------------------------------------------------------------------------
__global__ void zero_kernel(float* p, int n) {
  int i = blockIdx.x * blockDim.x + threadIdx.x;
  if (i < n) p[i] = 0.f;
}

__global__ void proto_accum_kernel(const float* __restrict__ feat,
                                   const int* __restrict__ lab,
                                   float* __restrict__ sums,
                                   float* __restrict__ cnt) {
  int i = blockIdx.x * blockDim.x + threadIdx.x;
  if (i >= B_SZ * D_DIM) return;
  int b = i >> 7, d = i & 127;
  int l = lab[b];
  atomicAdd(&sums[l * D_DIM + d], feat[i]);
  if (d == 0) atomicAdd(&cnt[l], 1.f);
}

__global__ void centers_kernel(const float* __restrict__ proto,
                               const float* __restrict__ sums,
                               const float* __restrict__ cnt,
                               float* __restrict__ cent,
                               unsigned int* __restrict__ cpack) {
  int c = blockIdx.x, d = threadIdx.x;  // grid CP=64, 128 threads
  if (c >= C_CLS) {                     // pad rows with zeros (bf16 B-fragments)
    if (d < D_DIM / 2) cpack[c * (D_DIM / 2) + d] = 0u;
    return;
  }
  float cc = cnt[c];
  float p = proto[c * D_DIM + d];
  float val = p;
  if (cc > 0.f) val = MOM * p + (1.f - MOM) * (sums[c * D_DIM + d] / fmaxf(cc, 1.f));
  __shared__ float red[128];
  red[d] = val * val;
  __syncthreads();
  for (int s = 64; s > 0; s >>= 1) {
    if (d < s) red[d] += red[d + s];
    __syncthreads();
  }
  float inv = 1.f / fmaxf(sqrtf(red[0]), 1e-12f);
  float nv = val * inv;
  cent[c * D_DIM + d] = nv;
  __shared__ float tmp[128];
  tmp[d] = nv;
  __syncthreads();
  if ((d & 1) == 0)
    cpack[c * (D_DIM / 2) + (d >> 1)] = pack2bf(tmp[d], tmp[d + 1]);
}

__global__ void ce_kernel(const float* __restrict__ cls,
                          const int* __restrict__ lab, float* __restrict__ ce) {
  int b = blockIdx.x * blockDim.x + threadIdx.x;
  if (b >= B_SZ) return;
  const float* row = cls + (size_t)b * C_CLS;
  float mx = -1e30f;
  for (int n = 0; n < C_CLS; ++n) mx = fmaxf(mx, row[n]);
  float se = 0.f;
  for (int n = 0; n < C_CLS; ++n) se += expf(row[n] - mx);
  atomicAdd(ce, (mx + logf(se)) - row[lab[b]]);
}

__global__ void topk_kernel(const float* __restrict__ scores,
                            float* __restrict__ tks, int* __restrict__ tki,
                            float* __restrict__ den) {
  __shared__ float s[N_TOK];
  int b = blockIdx.x, lane = threadIdx.x;  // 32 threads
  for (int i = lane; i < N_TOK; i += 32) s[i] = scores[(size_t)b * N_TOK + i];
  __syncthreads();
  float dsum = 0.f;
  for (int it = 0; it < K_TOP; ++it) {
    float bv = -1e30f;
    int bi = 0x7fffffff;
    for (int i = lane; i < N_TOK; i += 32) {
      float v = s[i];
      if (v > bv || (v == bv && i < bi)) { bv = v; bi = i; }
    }
    for (int off = 16; off > 0; off >>= 1) {
      float ov = __shfl_xor(bv, off, 32);
      int oi = __shfl_xor(bi, off, 32);
      if (ov > bv || (ov == bv && oi < bi)) { bv = ov; bi = oi; }
    }
    if (lane == 0) {
      tks[b * K_TOP + it] = bv;
      tki[b * K_TOP + it] = bi;
      s[bi] = -1e30f;
      dsum += bv;
    }
    __syncthreads();
  }
  if (lane == 0) den[b] = dsum;
}

// ---------------------------------------------------------------------------
// Kernel: token->prototype contrastive CE via WMMA.
// L2-norm applied post-WMMA (linearity), log-softmax via LDS per row.
// ---------------------------------------------------------------------------
__global__ __launch_bounds__(128) void contrast_kernel(
    const float* __restrict__ st, const unsigned int* __restrict__ cpack,
    const float* __restrict__ tks, const int* __restrict__ tki,
    const int* __restrict__ lab, float* __restrict__ num) {
  __shared__ float lg[WAVES][16 * CP];
  int tid = threadIdx.x, wv = tid >> 5, lane = tid & 31, lm = lane & 15,
      hf = lane >> 4;
  int tile = blockIdx.x * WAVES + wv;   // 8 tiles per batch row (128/16)
  int b = tile >> 3;
  int j = (tile & 7) * 16 + lm;
  int idx = tki[b * K_TOP + j];
  const float* xr = st + ((size_t)b * N_TOK + idx) * D_DIM;

  BFrag A[4];
  float ss = 0.f;
#pragma unroll
  for (int c = 0; c < 4; ++c)
#pragma unroll
    for (int v = 0; v < 8; ++v) {
      int k = c * 32 + hf * 16 + 2 * v;
      float2 p = *reinterpret_cast<const float2*>(xr + k);
      ss += p.x * p.x + p.y * p.y;
      A[c].u[v] = pack2bf(p.x, p.y);
    }
  ss += __shfl_xor(ss, 16, 32);
  float inv = 1.f / fmaxf(sqrtf(ss), 1e-12f);

  v8f acc[4];
#pragma unroll
  for (int t = 0; t < 4; ++t)
    acc[t] = (v8f){0.f, 0.f, 0.f, 0.f, 0.f, 0.f, 0.f, 0.f};
#pragma unroll
  for (int t = 0; t < 4; ++t)
#pragma unroll
    for (int c = 0; c < 4; ++c)
      acc[t] = wmma_bf16(
          A[c],
          bfrag_lds(cpack + (size_t)(t * 16 + lm) * (D_DIM / 2) + c * 16 + hf * 8),
          acc[t]);
#pragma unroll
  for (int t = 0; t < 4; ++t)
#pragma unroll
    for (int v = 0; v < 8; ++v) {
      int m = hf * 8 + v;
      float invm = __shfl(inv, m, 32);  // lane m holds row m's 1/norm
      lg[wv][m * CP + t * 16 + lm] = acc[t][v] * invm * TAU_INV;
    }
  __syncthreads();

  if (lane < 16) {
    const float* row = lg[wv] + lm * CP;
    float mx = -1e30f;
    for (int n = 0; n < C_CLS; ++n) mx = fmaxf(mx, row[n]);
    float se = 0.f;
    for (int n = 0; n < C_CLS; ++n) se += expf(row[n] - mx);
    float pt = (mx + logf(se)) - row[lab[b]];
    atomicAdd(&num[b], pt * tks[b * K_TOP + j]);
  }
}

__global__ void final_kernel(const float* __restrict__ num,
                             const float* __restrict__ den,
                             const float* __restrict__ ce,
                             float* __restrict__ loss) {
  __shared__ float red[256];
  int t = threadIdx.x;
  float s = 0.f;
  for (int b = t; b < B_SZ; b += 256) s += num[b] / den[b];
  red[t] = s;
  __syncthreads();
  for (int k = 128; k > 0; k >>= 1) {
    if (t < k) red[t] += red[t + k];
    __syncthreads();
  }
  if (t == 0)
    loss[0] = ce[0] * (1.f / B_SZ) + ALPHA * (red[0] * (1.f / B_SZ));
}

// ---------------------------------------------------------------------------
extern "C" void kernel_launch(void* const* d_in, const int* in_sizes, int n_in,
                              void* d_out, int out_size, void* d_ws,
                              size_t ws_size, hipStream_t stream) {
  (void)in_sizes; (void)n_in; (void)out_size; (void)ws_size;
  const float* x    = (const float*)d_in[0];
  const float* ct   = (const float*)d_in[1];
  const int*   lab  = (const int*)d_in[2];
  const float* w1   = (const float*)d_in[3];
  const float* b1   = (const float*)d_in[4];
  const float* g1   = (const float*)d_in[5];
  const float* be1  = (const float*)d_in[6];
  const float* w2   = (const float*)d_in[7];
  const float* b2   = (const float*)d_in[8];
  const float* wc1  = (const float*)d_in[9];
  const float* bc1  = (const float*)d_in[10];
  const float* gc   = (const float*)d_in[11];
  const float* bec  = (const float*)d_in[12];
  const float* wc2  = (const float*)d_in[13];
  const float* bc2  = (const float*)d_in[14];
  const float* wa1  = (const float*)d_in[15];
  const float* ba1  = (const float*)d_in[16];
  const float* wa2  = (const float*)d_in[17];
  const float* ba2  = (const float*)d_in[18];
  const float* prot = (const float*)d_in[19];

  float* ws = (float*)d_ws;
  float* st_ws   = ws;                                       // B*N*D
  float* sc_ws   = st_ws + (size_t)B_SZ * N_TOK * D_DIM;     // B*N
  float* feat_ws = sc_ws + (size_t)B_SZ * N_TOK;             // B*D
  float* tks_ws  = feat_ws + (size_t)B_SZ * D_DIM;           // B*K
  int*   tki_ws  = (int*)(tks_ws + (size_t)B_SZ * K_TOP);    // B*K
  float* den_ws  = (float*)(tki_ws + (size_t)B_SZ * K_TOP);  // B
  float* cent_ws = den_ws + B_SZ;                            // C*D
  unsigned int* cpack_ws = (unsigned int*)(cent_ws + C_CLS * D_DIM);  // CP*D/2
  float* zbase   = (float*)(cpack_ws + CP * (D_DIM / 2));
  float* sums_ws = zbase;                                    // C*D
  float* cnt_ws  = sums_ws + C_CLS * D_DIM;                  // 64
  float* num_ws  = cnt_ws + 64;                              // B
  float* ce_ws   = num_ws + B_SZ;                            // 1
  float* out     = (float*)d_out;

  int zn = C_CLS * D_DIM + 64 + B_SZ + 1;
  zero_kernel<<<(zn + 255) / 256, 256, 0, stream>>>(zbase, zn);

  size_t tok_lds = (size_t)(64 * 256 + 128 * 128) * 4 + 1024 * 4 +
                   (size_t)WAVES * 16 * H_DIM * 2;  // 164 KB
  token_mlp_kernel<<<(B_SZ * N_TOK) / (16 * WAVES), 128, tok_lds, stream>>>(
      x, w1, b1, g1, be1, w2, b2, wa1, ba1, wa2, ba2, st_ws, sc_ws);

  head_kernel<<<B_SZ / (16 * WAVES), 128, 0, stream>>>(
      ct, w1, b1, g1, be1, w2, b2, feat_ws, D_DIM, D_DIM, 8);
  head_kernel<<<B_SZ / (16 * WAVES), 128, 0, stream>>>(
      ct, wc1, bc1, gc, bec, wc2, bc2, out, C_CLS, C_CLS, 4);

  proto_accum_kernel<<<(B_SZ * D_DIM) / 256, 256, 0, stream>>>(feat_ws, lab,
                                                               sums_ws, cnt_ws);
  centers_kernel<<<CP, 128, 0, stream>>>(prot, sums_ws, cnt_ws, cent_ws,
                                         cpack_ws);
  ce_kernel<<<(B_SZ + 255) / 256, 256, 0, stream>>>(out, lab, ce_ws);
  topk_kernel<<<B_SZ, 32, 0, stream>>>(sc_ws, tks_ws, tki_ws, den_ws);
  contrast_kernel<<<(B_SZ * K_TOP / 16) / WAVES, 128, 0, stream>>>(
      st_ws, cpack_ws, tks_ws, tki_ws, lab, num_ws);
  final_kernel<<<1, 256, 0, stream>>>(num_ws, den_ws, ce_ws,
                                      out + (size_t)B_SZ * C_CLS);
}
